// MultiHeadAttention_77154792505388
// MI455X (gfx1250) — compile-verified, matrix operand order checked by reading, not run on hardware
//
#include <hip/hip_runtime.h>
#include <math.h>

// ---------------------------------------------------------------------------
// Types for CDNA5 WMMA (wave32, 16x16x32 bf16 -> f32)
// ---------------------------------------------------------------------------
typedef __bf16 bf16x16 __attribute__((ext_vector_type(16)));
typedef float  f32x8   __attribute__((ext_vector_type(8)));

#define D_MODEL 1024
#define N_HEADS 16
#define D_HEAD  64
#define SEQ_T   2048
#define NBATCH  2
#define ROWS    (NBATCH * SEQ_T)   // 4096

// ---------------------------------------------------------------------------
// bf16 helpers (round-to-nearest-even-ish, no reliance on native cvt)
// ---------------------------------------------------------------------------
__device__ inline __bf16 f2bf(float f) {
    union { float f; unsigned u; } a; a.f = f;
    unsigned r = a.u + 0x7FFFu + ((a.u >> 16) & 1u);
    union { unsigned short s; __bf16 b; } o; o.s = (unsigned short)(r >> 16);
    return o.b;
}
__device__ inline float bf2f(__bf16 b) {
    union { unsigned short s; __bf16 b; } i; i.b = b;
    union { unsigned u; float f; } o; o.u = ((unsigned)i.s) << 16;
    return o.f;
}

// Branchless grok softcap: 30*tanh(x/30) without libm's EXEC-masked range split.
// tanh(y) = (e^{2y}-1)/(e^{2y}+1); clamp keeps exp finite, result saturates ±30.
__device__ inline float softcap30(float x) {
    float y = x * (1.0f / 30.0f);
    y = fminf(fmaxf(y, -15.0f), 15.0f);
    float e = __expf(2.0f * y);
    return 30.0f * (e - 1.0f) / (e + 1.0f);
}

// ---------------------------------------------------------------------------
// Load a 16x32 bf16 A-style WMMA fragment from a row-major matrix.
// Per ISA 7.12.2 (16-bit A 16x32): lane L holds row m=L%16; half=L/16 selects
// K groups {half*8..+7} and {16+half*8..+7} -> two contiguous b128 loads.
// The B operand uses the same register layout applied to B^T rows, so all
// weight / K / V^T matrices are kept in "row-of-length-K" form.
// ---------------------------------------------------------------------------
__device__ inline bf16x16 load_frag_row(const __bf16* base, int ld) {
    int lane = threadIdx.x & 31;
    int m    = lane & 15;
    int hf   = lane >> 4;
    const __bf16* row = base + (size_t)m * ld + hf * 8;
    union { uint4 u[2]; bf16x16 v; } t;
    t.u[0] = *(const uint4*)(row);        // K = hf*8 .. hf*8+7
    t.u[1] = *(const uint4*)(row + 16);   // K = 16+hf*8 .. +7
    return t.v;
}

// ---------------------------------------------------------------------------
// Kernel 1: LayerNorm (f32 in) -> bf16 out, one 1024-wide row per block
// ---------------------------------------------------------------------------
__global__ void ln_bf16_kernel(const float* __restrict__ x,
                               const float* __restrict__ g,
                               const float* __restrict__ b,
                               __bf16* __restrict__ y) {
    int row = blockIdx.x;
    const float* xr = x + (size_t)row * D_MODEL;
    float s = 0.f, s2 = 0.f;
    for (int i = threadIdx.x; i < D_MODEL; i += 256) {
        float v = xr[i]; s += v; s2 += v * v;
    }
    __shared__ float red0[256], red1[256];
    red0[threadIdx.x] = s; red1[threadIdx.x] = s2;
    __syncthreads();
    for (int off = 128; off > 0; off >>= 1) {
        if (threadIdx.x < off) {
            red0[threadIdx.x] += red0[threadIdx.x + off];
            red1[threadIdx.x] += red1[threadIdx.x + off];
        }
        __syncthreads();
    }
    float mean = red0[0] * (1.0f / D_MODEL);
    float var  = red1[0] * (1.0f / D_MODEL) - mean * mean;
    float inv  = rsqrtf(var + 1e-5f);
    for (int i = threadIdx.x; i < D_MODEL; i += 256)
        y[(size_t)row * D_MODEL + i] = f2bf((xr[i] - mean) * inv * g[i] + b[i]);
}

// ---------------------------------------------------------------------------
// Kernel 2: weight convert + transpose:  Wt[n][k] = (bf16) W[k][n]   (1024x1024)
// ---------------------------------------------------------------------------
__global__ void wtrans_kernel(const float* __restrict__ W, __bf16* __restrict__ Wt) {
    int idx = blockIdx.x * 256 + threadIdx.x;   // over 1M elements
    int n = idx >> 10, k = idx & 1023;
    Wt[idx] = f2bf(W[(size_t)k * 1024 + n]);
}

// ---------------------------------------------------------------------------
// Kernel 3: generic bf16 WMMA GEMM:  C[M x N] = A[M x K] * Bt[N x K]^T
// One wave per block; each wave computes a 32x64 tile of C:
// 2 A-fragments + 4 B-fragments -> 8 WMMAs per 32-wide K step.
// ---------------------------------------------------------------------------
__device__ inline void store_out(float*  C, size_t off, float v) { C[off] = v; }
__device__ inline void store_out(__bf16* C, size_t off, float v) { C[off] = f2bf(v); }

template <typename OutT>
__global__ __launch_bounds__(32)
void gemm_bf16_kernel(const __bf16* __restrict__ A, const __bf16* __restrict__ Bt,
                      OutT* __restrict__ C, int M, int Nn, int K) {
    int lane = threadIdx.x & 31;
    int col0 = blockIdx.x * 64;
    int row0 = blockIdx.y * 32;
    f32x8 acc[2][4] = {};
    for (int k0 = 0; k0 < K; k0 += 32) {
        bf16x16 af0 = load_frag_row(A + (size_t)row0 * K + k0, K);
        bf16x16 af1 = load_frag_row(A + (size_t)(row0 + 16) * K + k0, K);
#pragma unroll
        for (int c = 0; c < 4; ++c) {
            bf16x16 bfv = load_frag_row(Bt + (size_t)(col0 + c * 16) * K + k0, K);
            acc[0][c] = __builtin_amdgcn_wmma_f32_16x16x32_bf16(
                false, af0, false, bfv, (short)0, acc[0][c], false, false);
            acc[1][c] = __builtin_amdgcn_wmma_f32_16x16x32_bf16(
                false, af1, false, bfv, (short)0, acc[1][c], false, false);
        }
    }
    int hf = lane >> 4, n = lane & 15;
#pragma unroll
    for (int mt = 0; mt < 2; ++mt)
#pragma unroll
        for (int c = 0; c < 4; ++c)
#pragma unroll
            for (int r = 0; r < 8; ++r) {
                size_t off = (size_t)(row0 + mt * 16 + r + 8 * hf) * Nn
                           + col0 + c * 16 + n;
                store_out(C, off, acc[mt][c][r]);
            }
}

// ---------------------------------------------------------------------------
// Kernel 4: RoPE + per-head relayout:  P[row][h*64+d] -> Qh[(b*16+h)*T + t][64]
// ---------------------------------------------------------------------------
__global__ void rope_heads_kernel(const __bf16* __restrict__ P, __bf16* __restrict__ Qh) {
    int row = blockIdx.x;                        // 0..4095
    int p   = blockIdx.y * 256 + threadIdx.x;    // 0..511 (pair index)
    int h = p >> 5, i = p & 31, d = 2 * i;
    int t = row & (SEQ_T - 1), batch = row >> 11;
    float ang = (float)t * __powf(10000.0f, -(float)d * (1.0f / 64.0f));
    float c = __cosf(ang), sn = __sinf(ang);
    size_t ib = (size_t)row * D_MODEL + h * 64 + d;
    float x0 = bf2f(P[ib]), x1 = bf2f(P[ib + 1]);
    size_t ob = ((size_t)(batch * N_HEADS + h) * SEQ_T + t) * 64 + d;
    Qh[ob]     = f2bf(x0 * c - x1 * sn);
    Qh[ob + 1] = f2bf(x1 * c + x0 * sn);
}

// ---------------------------------------------------------------------------
// Kernel 5: V relayout+transpose: P[row][h*64+d] -> Vt[(b*16+h)*64 + d][T]
// (so the PV B-fragment loads contiguously along T)
// ---------------------------------------------------------------------------
__global__ void vtrans_kernel(const __bf16* __restrict__ P, __bf16* __restrict__ Vt) {
    int row = blockIdx.x;
    int c   = blockIdx.y * 256 + threadIdx.x;    // 0..1023
    int h = c >> 6, d = c & 63;
    int t = row & (SEQ_T - 1), batch = row >> 11;
    Vt[((size_t)(batch * N_HEADS + h) * 64 + d) * SEQ_T + t] =
        P[(size_t)row * D_MODEL + c];
}

// ---------------------------------------------------------------------------
// Kernel 6: flash attention, one wave per (head, 16-query tile).
// S = Q K^T via WMMA, 1/sqrt(64) scale, branchless 30*tanh(x/30) softcap,
// causal+pad mask, online softmax (shfl_xor row-reductions within each
// 16-lane half), P re-layout through LDS (C-frag -> A-frag), PV via WMMA.
// ---------------------------------------------------------------------------
__global__ __launch_bounds__(32)
void attn_kernel(const __bf16* __restrict__ Qh, const __bf16* __restrict__ Kh,
                 const __bf16* __restrict__ Vt, const unsigned char* __restrict__ pad,
                 __bf16* __restrict__ Attn) {
    int lane = threadIdx.x & 31;
    int hf = lane >> 4, nl = lane & 15;
    int qt = blockIdx.x;          // query tile (16 rows)
    int bh = blockIdx.y;          // batch*16 + head
    int batch = bh >> 4, h = bh & 15;

    const __bf16* qbase = Qh + ((size_t)bh * SEQ_T + qt * 16) * 64;
    bf16x16 qf0 = load_frag_row(qbase, 64);        // d = 0..31
    bf16x16 qf1 = load_frag_row(qbase + 32, 64);   // d = 32..63

    float m_vec[8], l_vec[8];
    f32x8 o_acc[4] = {};
#pragma unroll
    for (int r = 0; r < 8; ++r) { m_vec[r] = -INFINITY; l_vec[r] = 0.f; }

    __shared__ __attribute__((aligned(16))) __bf16 pls[16 * 32];

    int jend = (qt + 1) * 16;
    for (int j0 = 0; j0 < jend; j0 += 32) {
        f32x8 s[2] = {};
#pragma unroll
        for (int sub = 0; sub < 2; ++sub) {
            const __bf16* kb = Kh + ((size_t)bh * SEQ_T + j0 + sub * 16) * 64;
            bf16x16 kf0 = load_frag_row(kb, 64);
            bf16x16 kf1 = load_frag_row(kb + 32, 64);
            s[sub] = __builtin_amdgcn_wmma_f32_16x16x32_bf16(
                false, qf0, false, kf0, (short)0, s[sub], false, false);
            s[sub] = __builtin_amdgcn_wmma_f32_16x16x32_bf16(
                false, qf1, false, kf1, (short)0, s[sub], false, false);
        }
        // scale + softcap + mask; track row max
        float p[2][8], rowmax[8];
#pragma unroll
        for (int r = 0; r < 8; ++r) rowmax[r] = -INFINITY;
#pragma unroll
        for (int sub = 0; sub < 2; ++sub) {
            int kn = j0 + sub * 16 + nl;
            bool padm = pad[batch * SEQ_T + kn] != 0;
#pragma unroll
            for (int r = 0; r < 8; ++r) {
                int qm = qt * 16 + r + 8 * hf;
                float sv = softcap30(s[sub][r] * 0.125f);    // 1/sqrt(64), cap
                if (padm || kn > qm) sv = -1e30f;            // pad | causal
                p[sub][r] = sv;
                rowmax[r] = fmaxf(rowmax[r], sv);
            }
        }
        // row max across the 16 lanes of each half (rows r and r+8)
#pragma unroll
        for (int r = 0; r < 8; ++r) {
#pragma unroll
            for (int mk = 1; mk < 16; mk <<= 1)
                rowmax[r] = fmaxf(rowmax[r], __shfl_xor(rowmax[r], mk, 32));
        }
        float alpha[8], rowsum[8];
#pragma unroll
        for (int r = 0; r < 8; ++r) {
            float mn = fmaxf(m_vec[r], rowmax[r]);
            alpha[r] = __expf(m_vec[r] - mn);
            m_vec[r] = mn;
            float rs = 0.f;
#pragma unroll
            for (int sub = 0; sub < 2; ++sub) {
                float pv = __expf(p[sub][r] - mn);
                p[sub][r] = pv;
                rs += pv;
            }
            rowsum[r] = rs;
        }
#pragma unroll
        for (int r = 0; r < 8; ++r) {
#pragma unroll
            for (int mk = 1; mk < 16; mk <<= 1)
                rowsum[r] += __shfl_xor(rowsum[r], mk, 32);
            l_vec[r] = l_vec[r] * alpha[r] + rowsum[r];
        }
#pragma unroll
        for (int c = 0; c < 4; ++c)
#pragma unroll
            for (int r = 0; r < 8; ++r) o_acc[c][r] = o_acc[c][r] * alpha[r];

        // P: C-fragment layout -> LDS (16x32 bf16) -> A-fragment layout
        __syncthreads();
#pragma unroll
        for (int sub = 0; sub < 2; ++sub)
#pragma unroll
            for (int r = 0; r < 8; ++r)
                pls[(r + 8 * hf) * 32 + sub * 16 + nl] = f2bf(p[sub][r]);
        __syncthreads();
        bf16x16 pf = load_frag_row(pls, 32);

        // O += P(16x32) * V(32x64): 4 column tiles from V^T rows
#pragma unroll
        for (int c = 0; c < 4; ++c) {
            const __bf16* vb = Vt + ((size_t)bh * 64 + c * 16) * SEQ_T + j0;
            bf16x16 vf = load_frag_row(vb, SEQ_T);
            o_acc[c] = __builtin_amdgcn_wmma_f32_16x16x32_bf16(
                false, pf, false, vf, (short)0, o_acc[c], false, false);
        }
    }
    // epilogue: O /= l ; write to Attn[N][T][H*64] (bf16, A of final GEMM)
#pragma unroll
    for (int c = 0; c < 4; ++c)
#pragma unroll
        for (int r = 0; r < 8; ++r) {
            float v = o_acc[c][r] / l_vec[r];
            int row = qt * 16 + r + 8 * hf;
            size_t off = ((size_t)batch * SEQ_T + row) * D_MODEL + h * 64 + c * 16 + nl;
            Attn[off] = f2bf(v);
        }
}

// ---------------------------------------------------------------------------
// Host-side launch.  Workspace layout (bf16 buffers, ~88 MB total):
//   xqn/xkn/xvn (3 x 8MB) | wqt/wkt/wvt/wot (4 x 2MB) |
//   pq/pk/pv (3 x 8MB) | qh/kh (2 x 8MB) | vt (8MB) | attn (8MB)
// ---------------------------------------------------------------------------
extern "C" void kernel_launch(void* const* d_in, const int* in_sizes, int n_in,
                              void* d_out, int out_size, void* d_ws, size_t ws_size,
                              hipStream_t stream) {
    const float* xq  = (const float*)d_in[0];
    const float* xk  = (const float*)d_in[1];
    const float* xv  = (const float*)d_in[2];
    const unsigned char* pad = (const unsigned char*)d_in[3];
    const float* Wq  = (const float*)d_in[4];
    const float* Wk  = (const float*)d_in[5];
    const float* Wv  = (const float*)d_in[6];
    const float* Wo  = (const float*)d_in[7];
    const float* lng = (const float*)d_in[8];
    const float* lnb = (const float*)d_in[9];
    float* out = (float*)d_out;

    char* ws = (char*)d_ws;
    const size_t RB = (size_t)ROWS * D_MODEL * sizeof(__bf16);    // 8 MB
    const size_t WB = (size_t)D_MODEL * D_MODEL * sizeof(__bf16); // 2 MB
    __bf16* xqn  = (__bf16*)(ws);
    __bf16* xkn  = (__bf16*)(ws + RB);
    __bf16* xvn  = (__bf16*)(ws + 2 * RB);
    __bf16* wqt  = (__bf16*)(ws + 3 * RB);
    __bf16* wkt  = (__bf16*)(ws + 3 * RB + WB);
    __bf16* wvt  = (__bf16*)(ws + 3 * RB + 2 * WB);
    __bf16* wot  = (__bf16*)(ws + 3 * RB + 3 * WB);
    __bf16* pq   = (__bf16*)(ws + 3 * RB + 4 * WB);
    __bf16* pk   = (__bf16*)(ws + 4 * RB + 4 * WB);
    __bf16* pv   = (__bf16*)(ws + 5 * RB + 4 * WB);
    __bf16* qh   = (__bf16*)(ws + 6 * RB + 4 * WB);
    __bf16* kh   = (__bf16*)(ws + 7 * RB + 4 * WB);
    __bf16* vt   = (__bf16*)(ws + 8 * RB + 4 * WB);
    __bf16* attn = (__bf16*)(ws + 9 * RB + 4 * WB);

    // 1) LayerNorm -> bf16
    ln_bf16_kernel<<<ROWS, 256, 0, stream>>>(xq, lng, lnb, xqn);
    ln_bf16_kernel<<<ROWS, 256, 0, stream>>>(xk, lng, lnb, xkn);
    ln_bf16_kernel<<<ROWS, 256, 0, stream>>>(xv, lng, lnb, xvn);

    // 2) Weight convert + transpose
    wtrans_kernel<<<4096, 256, 0, stream>>>(Wq, wqt);
    wtrans_kernel<<<4096, 256, 0, stream>>>(Wk, wkt);
    wtrans_kernel<<<4096, 256, 0, stream>>>(Wv, wvt);
    wtrans_kernel<<<4096, 256, 0, stream>>>(Wo, wot);

    // 3) QKV projections (WMMA, 32x64 tile per wave)
    dim3 ggrid(D_MODEL / 64, ROWS / 32);   // 16 x 128
    gemm_bf16_kernel<__bf16><<<ggrid, 32, 0, stream>>>(xqn, wqt, pq, ROWS, D_MODEL, D_MODEL);
    gemm_bf16_kernel<__bf16><<<ggrid, 32, 0, stream>>>(xkn, wkt, pk, ROWS, D_MODEL, D_MODEL);
    gemm_bf16_kernel<__bf16><<<ggrid, 32, 0, stream>>>(xvn, wvt, pv, ROWS, D_MODEL, D_MODEL);

    // 4) RoPE + head relayout for Q and K; transpose relayout for V
    rope_heads_kernel<<<dim3(ROWS, 2), 256, 0, stream>>>(pq, qh);
    rope_heads_kernel<<<dim3(ROWS, 2), 256, 0, stream>>>(pk, kh);
    vtrans_kernel<<<dim3(ROWS, 4), 256, 0, stream>>>(pv, vt);

    // 5) Flash attention (WMMA + LDS relayout + online softmax)
    attn_kernel<<<dim3(SEQ_T / 16, NBATCH * N_HEADS), 32, 0, stream>>>(qh, kh, vt, pad, attn);

    // 6) Output projection (WMMA, f32 out)
    gemm_bf16_kernel<float><<<ggrid, 32, 0, stream>>>(attn, wot, out, ROWS, D_MODEL, D_MODEL);
}
